// TransformerModule_36103495090684
// MI455X (gfx1250) — matrix operand order repaired and checked by reading, not compiled, not run on hardware
//
#include <hip/hip_runtime.h>
#include <math.h>

// ---------------------------------------------------------------------------
// Sizes from the reference: B=64, F=512, E=256, H=8, L=4, D=32
// ---------------------------------------------------------------------------
#define Bsz 64
#define Fsz 512
#define Esz 256
#define Hsz 8
#define Lsz 4
#define Dsz 32

typedef __attribute__((ext_vector_type(16))) _Float16    v16h;
typedef __attribute__((ext_vector_type(8)))  float       v8f;
typedef __attribute__((ext_vector_type(8)))  unsigned int v8u;

union FragU {
  unsigned int a[8];  // addressable view for LDS memcpy
  v8u          u;
  v16h         h;
};

__device__ __forceinline__ unsigned int pkf16(float a, float b) {
  union { _Float16 h[2]; unsigned int u; } t;
  t.h[0] = (_Float16)a;
  t.h[1] = (_Float16)b;
  return t.u;  // v_cvt_pk_f16_f32
}

// Identity half-swap between lanes 0-15 and 16-31 (pure VALU, no LDS).
__device__ __forceinline__ unsigned int permx16u(unsigned int v) {
  return (unsigned int)__builtin_amdgcn_permlanex16(
      (int)v, (int)v, 0x76543210, 0xFEDCBA98, false, false);
}
__device__ __forceinline__ float permx16f(float v) {
  int i = __builtin_bit_cast(int, v);
  i = __builtin_amdgcn_permlanex16(i, i, 0x76543210, 0xFEDCBA98, false, false);
  return __builtin_bit_cast(float, i);
}

__device__ __forceinline__ v8f wmma_f16(v16h a, v16h b, v8f c) {
  return __builtin_amdgcn_wmma_f32_16x16x32_f16(false, a, false, b, (short)0, c,
                                                false, false);
}

__device__ __forceinline__ float2 ldf2(const float* p) {
  float2 r;
  __builtin_memcpy(&r, (const void*)__builtin_assume_aligned(p, 8), 8);
  return r;
}

// ---------------------------------------------------------------------------
// LDS fragment cache: per (chunk c, frag fr) a region of 32 lanes x 32 bytes.
// The two 16B sub-blocks of a lane are slot-swizzled by (lane>>3)&1 so each
// ds_load_b128 request spreads over both slot parities (2-way instead of
// 4-way bank conflicts). Store and load use the same per-lane mapping.
// ---------------------------------------------------------------------------
__device__ __forceinline__ void stfrag(const FragU& f, _Float16* region,
                                       int lane) {
  const int sA = (lane >> 3) & 1;
  __builtin_memcpy(
      (void*)__builtin_assume_aligned(region + (lane * 2 + sA) * 8, 16),
      &f.a[0], 16);
  __builtin_memcpy(
      (void*)__builtin_assume_aligned(region + (lane * 2 + 1 - sA) * 8, 16),
      &f.a[4], 16);
}
__device__ __forceinline__ v16h ldfrag(const _Float16* region, int lane) {
  FragU f;
  const int sA = (lane >> 3) & 1;
  __builtin_memcpy(
      &f.a[0],
      (const void*)__builtin_assume_aligned(region + (lane * 2 + sA) * 8, 16),
      16);
  __builtin_memcpy(&f.a[4],
                   (const void*)__builtin_assume_aligned(
                       region + (lane * 2 + 1 - sA) * 8, 16),
                   16);
  return f.h;
}

// Build a 32x16 f16 B-fragment from two C-layout f32 tiles (half-swap only).
__device__ __forceinline__ v16h make_B(v8f t0, v8f t1, int hi) {
  unsigned int pk0[4], pk1[4];
#pragma unroll
  for (int j = 0; j < 4; ++j) {
    pk0[j] = pkf16(t0[2 * j], t0[2 * j + 1]);
    pk1[j] = pkf16(t1[2 * j], t1[2 * j + 1]);
  }
  FragU f;
#pragma unroll
  for (int j = 0; j < 4; ++j) {
    unsigned int s0 = permx16u(pk0[j]);
    unsigned int s1 = permx16u(pk1[j]);
    f.u[j]     = hi ? s1 : pk0[j];
    f.u[4 + j] = hi ? pk1[j] : s0;
  }
  return f.h;
}

// ---------------------------------------------------------------------------
// Prep kernel: rank-1 embedding collapse. One block, 256 threads.
// ws (floats): [0)kv [256)kc [512)vv [768)vc [1024)qv
// ---------------------------------------------------------------------------
__global__ void prep_kernel(const float* __restrict__ emb_w,
                            const float* __restrict__ emb_b,
                            const float* __restrict__ q_w,
                            const float* __restrict__ q_b,
                            const float* __restrict__ k_w,
                            const float* __restrict__ k_b,
                            const float* __restrict__ v_w,
                            const float* __restrict__ v_b,
                            float* __restrict__ ws) {
  int e = threadIdx.x;
  float kv = 0.f, kc = 0.f, vv = 0.f, vc = 0.f, qv = 0.f;
  for (int i = 0; i < Esz; ++i) {
    float ew = emb_w[i], eb = emb_b[i];
    float kwe = k_w[i * Esz + e];
    float vwe = v_w[i * Esz + e];
    kv = fmaf(ew, kwe, kv);
    kc = fmaf(eb, kwe, kc);
    vv = fmaf(ew, vwe, vv);
    vc = fmaf(eb, vwe, vc);
    qv = fmaf(eb, q_w[i * Esz + e], qv);
  }
  ws[e]           = kv;
  ws[Esz + e]     = kc + k_b[e];
  ws[2 * Esz + e] = vv;
  ws[3 * Esz + e] = vc + v_b[e];
  ws[4 * Esz + e] = qv + q_b[e];
}

__global__ void init_out_kernel(float* __restrict__ out,
                                const float* __restrict__ attn_b) {
  int i = blockIdx.x * 256 + threadIdx.x;
  if (i < 2 * Bsz * Fsz) out[i] = attn_b[0];
}

// ---------------------------------------------------------------------------
// Dynamic LDS layout (bytes):
//   [0, 65536)        sF   : fragment cache, 16 chunks x 4 frags x 32 lanes x 32B
//   [65536, 67584)    sX   : features row (512 f32)
//   [67584, ...)      consts: kv,kc,vv,vc,aw,qv (32 f32 each) + vsd
// ---------------------------------------------------------------------------
#define SMEM_BYTES (65536 + 2048 + 7 * 32 * 4)

__global__ __launch_bounds__(256, 2) void attn_kernel(
    const float* __restrict__ x,       // [B,F]
    const float* __restrict__ ws,      // prep output
    const float* __restrict__ attn_w,  // [E]
    float* __restrict__ out) {         // [2*B*F] : h_ox then h_c
  extern __shared__ char smem[];
  _Float16* sF = (_Float16*)smem;
  float* sX  = (float*)(smem + 65536);
  float* sKV = (float*)(smem + 65536 + 2048);
  float* sKC = sKV + 32;
  float* sVV = sKV + 64;
  float* sVC = sKV + 96;
  float* sAW = sKV + 128;
  float* sQV = sKV + 160;
  float* sVD = sKV + 192;

  const int tid  = threadIdx.x;
  const int lane = tid & 31;
  const int wave = tid >> 5;
  const int b = blockIdx.x >> 3;
  const int h = blockIdx.x & 7;
  const int m  = lane & 15;
  const int hi = lane >> 4;

  if (tid < Dsz) {
    int e = h * Dsz + tid;
    sKV[tid] = ws[e];
    sKC[tid] = ws[Esz + e];
    sVV[tid] = ws[2 * Esz + e];
    sVC[tid] = ws[3 * Esz + e];
    sQV[tid] = ws[4 * Esz + e];
    sAW[tid] = attn_w[e];
  }
  for (int f = tid; f < Fsz; f += 256) sX[f] = x[b * Fsz + f];
  __syncthreads();

  if (wave == 0) {
    float p = 0.f;
    for (int f = lane; f < Fsz; f += 32) p += sX[f];
#pragma unroll
    for (int msk = 16; msk; msk >>= 1) p += __shfl_xor(p, msk, 32);
    if (lane == 0) {
      float vd = 0.f;  // v_sum . attn_w for this head
      for (int d = 0; d < Dsz; ++d)
        vd += fmaf(p, sVV[d], (float)Fsz * sVC[d]) * sAW[d];
      sVD[0] = vd;
    }
  }

  // --- Build the fragment cache: wave w builds chunks 2w, 2w+1 -------------
  // K constants in A-layout k-order, pre-scaled by log2(e)/sqrt(D) so the
  // whole softmax runs in the log2 domain (native v_exp_f32).
  {
    const float s2 = 0.17677669529663687f * 1.4426950408889634f;
    float KVa[16], KCa[16];
#pragma unroll
    for (int e = 0; e < 16; ++e) {
      int k = e + 16 * hi;
      KVa[e] = sKV[k] * s2;
      KCa[e] = sKC[k] * s2;
    }
    const float vv0 = sVV[m], vc0 = sVC[m];
    const float vv1 = sVV[16 + m], vc1 = sVC[16 + m];
#pragma unroll
    for (int cc = 0; cc < 2; ++cc) {
      const int c = wave * 2 + cc;
      const int g0 = c * 32;
      const float xk0 = sX[g0 + m];
      const float xk1 = sX[g0 + 16 + m];
      FragU ak0, ak1, av0, av1;
#pragma unroll
      for (int j = 0; j < 8; ++j) {
        ak0.u[j] = pkf16(fmaf(xk0, KVa[2 * j], KCa[2 * j]),
                         fmaf(xk0, KVa[2 * j + 1], KCa[2 * j + 1]));
        ak1.u[j] = pkf16(fmaf(xk1, KVa[2 * j], KCa[2 * j]),
                         fmaf(xk1, KVa[2 * j + 1], KCa[2 * j + 1]));
        float2 xp = ldf2(&sX[g0 + 2 * j + 16 * hi]);
        av0.u[j] = pkf16(fmaf(xp.x, vv0, vc0), fmaf(xp.y, vv0, vc0));
        av1.u[j] = pkf16(fmaf(xp.x, vv1, vc1), fmaf(xp.y, vv1, vc1));
      }
      stfrag(ak0, sF + (c * 4 + 0) * 512, lane);
      stfrag(ak1, sF + (c * 4 + 1) * 512, lane);
      stfrag(av0, sF + (c * 4 + 2) * 512, lane);
      stfrag(av1, sF + (c * 4 + 3) * 512, lane);
    }
  }
  __syncthreads();

  // Layer-0 Q^T B-fragment: B[k=d][n=f] = qv[d], element e -> d = e+16*hi
  v16h Qb0;
#pragma unroll
  for (int e = 0; e < 16; ++e) Qb0[e] = (_Float16)sQV[e + 16 * hi];
  // attn_w in O^T C-layout order: OT0 vgpr i -> d=i+8*hi ; OT1 -> d=16+i+8*hi
  float awA[8], awB[8];
#pragma unroll
  for (int i = 0; i < 8; ++i) {
    awA[i] = sAW[i + 8 * hi];
    awB[i] = sAW[16 + i + 8 * hi];
  }
  const float vsd = sVD[0];

  for (int t = 0; t < 4; ++t) {            // 4 query-column tiles per wave
    const int f0 = (wave * 4 + t) * 16;
    v16h Qb = Qb0;
    for (int layer = 0; layer < Lsz; ++layer) {
      v8f OT0 = {}, OT1 = {};
      float M = -3.0e38f, L = 0.f;

      for (int c = 0; c < Fsz / 32; ++c) {  // 16 key chunks of 32
        // --- cached K / V^T A-fragments: 8x ds_load_b128 total ---
        v16h ak0 = ldfrag(sF + (c * 4 + 0) * 512, lane);
        v16h ak1 = ldfrag(sF + (c * 4 + 1) * 512, lane);
        v8f z = {};
        v8f S0 = wmma_f16(ak0, Qb, z);   // log2-domain S^T rows g0..g0+15
        v8f S1 = wmma_f16(ak1, Qb, z);   // log2-domain S^T rows g0+16..g0+31

        // --- per-lane online softmax (query f = lane column), base 2 ---
        float mx = fmaxf(S0[0], S1[0]);
#pragma unroll
        for (int i = 1; i < 8; ++i) mx = fmaxf(mx, fmaxf(S0[i], S1[i]));
        mx = fmaxf(mx, permx16f(mx));       // combine g-halves of same f
        const float nM = fmaxf(M, mx);
        const float al = __builtin_amdgcn_exp2f(M - nM);
        M = nM;
        v8f P0, P1;
        float rs = 0.f;
#pragma unroll
        for (int i = 0; i < 8; ++i) {
          float p0 = __builtin_amdgcn_exp2f(S0[i] - nM);
          float p1 = __builtin_amdgcn_exp2f(S1[i] - nM);
          P0[i] = p0;
          P1[i] = p1;
          rs += p0 + p1;
        }
        L = L * al + (rs + permx16f(rs));
#pragma unroll
        for (int i = 0; i < 8; ++i) {
          OT0[i] *= al;
          OT1[i] *= al;
        }
        // --- P^T as B fragment (half-swap only) ---
        v16h Bp = make_B(P0, P1, hi);
        v16h av0 = ldfrag(sF + (c * 4 + 2) * 512, lane);
        v16h av1 = ldfrag(sF + (c * 4 + 3) * 512, lane);
        OT0 = wmma_f16(av0, Bp, OT0);    // O^T rows d=0..15
        OT1 = wmma_f16(av1, Bp, OT1);    // O^T rows d=16..31
      }

      const float r = 1.0f / L;
      if (layer < Lsz - 1) {
        v8f q0, q1;
#pragma unroll
        for (int i = 0; i < 8; ++i) {
          q0[i] = OT0[i] * r;
          q1[i] = OT1[i] * r;
        }
        Qb = make_B(q0, q1, hi);           // Q^T for next layer, no LDS
      } else {
        float part = 0.f;
#pragma unroll
        for (int i = 0; i < 8; ++i)
          part += OT0[i] * awA[i] + OT1[i] * awB[i];
        part *= r;
        float tot = part + permx16f(part);  // both d-halves
        if (hi == 0) {
          int f = f0 + m;
          atomicAdd(&out[b * Fsz + f], tot);                    // h_ox
          atomicAdd(&out[Bsz * Fsz + b * Fsz + f], vsd - tot);  // h_c
        }
      }
    }
  }
}

// ---------------------------------------------------------------------------
extern "C" void kernel_launch(void* const* d_in, const int* in_sizes, int n_in,
                              void* d_out, int out_size, void* d_ws,
                              size_t ws_size, hipStream_t stream) {
  const float* features = (const float*)d_in[0];
  const float* emb_w    = (const float*)d_in[1];
  const float* emb_b    = (const float*)d_in[2];
  const float* q_w      = (const float*)d_in[3];
  const float* q_b      = (const float*)d_in[4];
  const float* k_w      = (const float*)d_in[5];
  const float* k_b      = (const float*)d_in[6];
  const float* v_w      = (const float*)d_in[7];
  const float* v_b      = (const float*)d_in[8];
  const float* attn_w   = (const float*)d_in[9];
  const float* attn_b   = (const float*)d_in[10];
  float* out = (float*)d_out;
  float* ws  = (float*)d_ws;

  prep_kernel<<<1, 256, 0, stream>>>(emb_w, emb_b, q_w, q_b, k_w, k_b, v_w,
                                     v_b, ws);
  init_out_kernel<<<(2 * Bsz * Fsz + 255) / 256, 256, 0, stream>>>(out, attn_b);
  attn_kernel<<<Bsz * Hsz, 256, SMEM_BYTES, stream>>>(features, ws, attn_w,
                                                      out);
}
